// ScopeMultiDNN_19533511262560
// MI455X (gfx1250) — matrix operand order repaired and checked by reading, not compiled, not run on hardware
//
#include <hip/hip_runtime.h>
#include <hip/hip_bf16.h>
#include <stdint.h>

// ---------------------------------------------------------------------------
// CDNA5 (gfx1250) WMMA types
// ---------------------------------------------------------------------------
typedef __attribute__((ext_vector_type(16))) __bf16 v16bf;
typedef __attribute__((ext_vector_type(8)))  float  v8f;

union Frag16 {
  v16bf h;
  uint4 q[2];
};

__device__ __forceinline__ uint16_t f32_to_bf16_rte(float f) {
  union { float f; uint32_t u; } v; v.f = f;
  uint32_t u = v.u;
  u += 0x7FFFu + ((u >> 16) & 1u);           // round-to-nearest-even
  return (uint16_t)(u >> 16);
}
__device__ __forceinline__ float bf16_to_f32(uint16_t h) {
  union { uint32_t u; float f; } v; v.u = ((uint32_t)h) << 16;
  return v.f;
}
__device__ __forceinline__ int rev8(int i) {
  i = ((i & 0x0F) << 4) | ((i >> 4) & 0x0F);
  i = ((i & 0x33) << 2) | ((i >> 2) & 0x33);
  i = ((i & 0x55) << 1) | ((i >> 1) & 0x55);
  return i;
}

// ---------------------------------------------------------------------------
// fp32 -> bf16 with zero padding (Rin x Cin) -> (Rout x Cout), Rout*Cout=total
// ---------------------------------------------------------------------------
__global__ void convert_pad_bf16(const float* __restrict__ src, int Rin, int Cin,
                                 uint16_t* __restrict__ dst, long total, int Cout) {
  long stride = (long)gridDim.x * blockDim.x;
  for (long idx = (long)blockIdx.x * blockDim.x + threadIdx.x; idx < total; idx += stride) {
    int r = (int)(idx / Cout);
    int c = (int)(idx % Cout);
    float v = (r < Rin && c < Cin) ? src[(long)r * Cin + c] : 0.0f;
    dst[idx] = f32_to_bf16_rte(v);
  }
}

// ---------------------------------------------------------------------------
// Register-blocked bf16 WMMA GEMM:  C[m,n] = act( sum_k A[m,k]*B[n,k] + bias[n] )
//   (C = A @ B^T, B row-major = weight matrix)
//
// Wave tile   : 32x64  (MT=2 x NT=4 tiles of 16x16) -> 8 WMMA per 32-K chunk
//               from 12 b128 loads (vs 1 WMMA : 4 loads unblocked).
// Block tile  : 8 waves (2x4) -> 64 x 256.
// A-frag : lane l (m=l&15, hh=l>>4): K = hh*8..+7 and 16+hh*8..+7  (2 b128)
// B-frag : lane l (n=l&15, kh=l>>4): K = kh*8..+7 and 16+kh*8..+7  (2 b128)
// C tile : VGPR r, lane l: row = r + 8*(l>>4), col = l&15
// act: 0=none 1=relu 2=leaky(0.01)
// ---------------------------------------------------------------------------
#define MT 2
#define NT 4

__global__ __launch_bounds__(256)
void gemm_bf16_wmma(const uint16_t* __restrict__ A, int lda,
                    const uint16_t* __restrict__ B, int ldb,
                    const float* __restrict__ bias, int bias_n,
                    uint16_t* __restrict__ Cb, float* __restrict__ Cf, int ldc,
                    int Npad, int K, int act) {
  const int wave = threadIdx.x >> 5;
  const int lane = threadIdx.x & 31;
  const int wm = wave >> 2;                       // 0..1
  const int wn = wave & 3;                        // 0..3
  const int mbase = blockIdx.x * (MT * 16 * 2) + wm * (MT * 16);   // 64/block
  const int nbase = blockIdx.y * (NT * 16 * 4) + wn * (NT * 16);   // 256/block
  if (nbase >= Npad) return;                      // wave-uniform: EXEC stays all-1s
  const int lo16 = lane & 15;
  const int hh   = lane >> 4;

  const uint16_t* arow[MT];
  const uint16_t* brow[NT];
#pragma unroll
  for (int i = 0; i < MT; ++i)
    arow[i] = A + (size_t)(mbase + i * 16 + lo16) * lda;
#pragma unroll
  for (int j = 0; j < NT; ++j)
    brow[j] = B + (size_t)(nbase + j * 16 + lo16) * ldb;

  v8f c[MT][NT] = {};

  for (int k0 = 0; k0 < K; k0 += 32) {
    Frag16 af[MT], bf[NT];
#pragma unroll
    for (int i = 0; i < MT; ++i) {
      af[i].q[0] = *(const uint4*)(arow[i] + k0 + hh * 8);
      af[i].q[1] = *(const uint4*)(arow[i] + k0 + 16 + hh * 8);
    }
#pragma unroll
    for (int j = 0; j < NT; ++j) {
      bf[j].q[0] = *(const uint4*)(brow[j] + k0 + hh * 8);
      bf[j].q[1] = *(const uint4*)(brow[j] + k0 + 16 + hh * 8);
    }
#pragma unroll
    for (int i = 0; i < MT; ++i)
#pragma unroll
      for (int j = 0; j < NT; ++j)
        c[i][j] = __builtin_amdgcn_wmma_f32_16x16x32_bf16(false, af[i].h, false, bf[j].h,
                                                          (short)0, c[i][j], false, false);
    __builtin_prefetch(arow[0] + k0 + 128, 0, 1);    // global_prefetch upcoming chunks
    __builtin_prefetch(brow[0] + k0 + 128, 0, 1);
  }

#pragma unroll
  for (int j = 0; j < NT; ++j) {
    const int nn = nbase + j * 16 + lo16;
    const float badd = (bias != nullptr && nn < bias_n) ? bias[nn] : 0.0f;
#pragma unroll
    for (int i = 0; i < MT; ++i) {
#pragma unroll
      for (int r = 0; r < 8; ++r) {
        int mm = mbase + i * 16 + r + (hh << 3);
        float v = c[i][j][r] + badd;
        if (act == 1)      v = fmaxf(v, 0.0f);
        else if (act == 2) v = v > 0.0f ? v : 0.01f * v;
        size_t o = (size_t)mm * ldc + nn;
        if (Cb) Cb[o] = f32_to_bf16_rte(v);
        if (Cf) Cf[o] = v;
      }
    }
  }
}

// ---------------------------------------------------------------------------
// Initial bit-reversal permutation: Af[b,i,c] = P0[b, rev8(i), c]
// Af, P0 : 4096 x 128 f32 (cols 120..127 are zero in P0 by construction)
// ---------------------------------------------------------------------------
__global__ void permute_init(const float* __restrict__ P0, float* __restrict__ Af) {
  int idx = blockIdx.x * blockDim.x + threadIdx.x;   // 4096*128 threads
  int r = idx >> 7, c = idx & 127;
  int b = r >> 8, i = r & 255;
  Af[idx] = P0[(size_t)((b << 8) + rev8(i)) * 128 + c];
}

// ---------------------------------------------------------------------------
// Butterfly pack: build op_in (2048 x 256 bf16, interleaved ax/ay, zero pad)
// row r -> b=r>>7, g=(r&127)>>stage, s=(r&127)&(stride-1)
// xrow = b*256 + g*2*stride + s ; yrow = xrow + stride
// op_in[2p]=a[xrow,p], op_in[2p+1]=a[yrow,p]
// ---------------------------------------------------------------------------
__global__ void butterfly_pack(const float* __restrict__ Af, uint16_t* __restrict__ OP,
                               int stage) {
  int idx = blockIdx.x * blockDim.x + threadIdx.x;   // 2048*256 threads
  int r = idx >> 8, c = idx & 255;
  int stride = 1 << stage;
  int b = r >> 7, rem = r & 127;
  int g = rem >> stage, s = rem & (stride - 1);
  int xrow = (b << 8) + (g << (stage + 1)) + s;
  float v = 0.0f;
  if (c < 240) {
    int p = c >> 1, half = c & 1;
    v = Af[(size_t)(xrow + half * stride) * 128 + p];
  }
  OP[idx] = f32_to_bf16_rte(v);
}

// ---------------------------------------------------------------------------
// Butterfly unpack: scatter h (2048 x 256 f32) back:
// nx = h[:,:120] -> xrow ; ny = h[:,120:240] -> yrow  (pad cols untouched)
// ---------------------------------------------------------------------------
__global__ void butterfly_unpack(const float* __restrict__ OUTf, float* __restrict__ Af,
                                 int stage) {
  int idx = blockIdx.x * blockDim.x + threadIdx.x;   // 2048*256 threads
  int r = idx >> 8, c = idx & 255;
  if (c >= 240) return;
  int stride = 1 << stage;
  int b = r >> 7, rem = r & 127;
  int g = rem >> stage, s = rem & (stride - 1);
  int xrow = (b << 8) + (g << (stage + 1)) + s;
  int half = (c < 120) ? 0 : 1;
  int p = (c < 120) ? c : (c - 120);
  Af[(size_t)(xrow + half * stride) * 128 + p] = OUTf[idx];
}

// ---------------------------------------------------------------------------
// Final bit-reversal gather: t_pitch[b*256+i, p] = Af[b*256 + rev8(i), p]
// ---------------------------------------------------------------------------
__global__ void final_gather(const float* __restrict__ Af, float* __restrict__ out) {
  int row = blockIdx.x;            // 0..4095
  int p = threadIdx.x;             // block 128
  if (p >= 120) return;
  int b = row >> 8, i = row & 255;
  out[(size_t)row * 120 + p] = Af[(size_t)((b << 8) + rev8(i)) * 128 + p];
}

// ---------------------------------------------------------------------------
// Voice head: out[row, j] = relu( sum_k H3[row,k]*Wv[j,k] + bv[j] ), j in {0,1}
// ---------------------------------------------------------------------------
__global__ void voice_head(const uint16_t* __restrict__ H3, const float* __restrict__ Wv,
                           const float* __restrict__ bvv, float* __restrict__ out) {
  int t = blockIdx.x * blockDim.x + threadIdx.x;     // 8192 threads
  int row = t >> 1, j = t & 1;
  const uint16_t* h = H3 + (size_t)row * 1024;
  const float* w = Wv + j * 1024;
  float acc = bvv[j];
  for (int k = 0; k < 1024; ++k) acc += bf16_to_f32(h[k]) * w[k];
  out[(size_t)row * 2 + j] = fmaxf(acc, 0.0f);
}

// ---------------------------------------------------------------------------
// Host launcher
// ---------------------------------------------------------------------------
extern "C" void kernel_launch(void* const* d_in, const int* in_sizes, int n_in,
                              void* d_out, int out_size, void* d_ws, size_t ws_size,
                              hipStream_t stream) {
  const float* feat = (const float*)d_in[0];
  const float* W0 = (const float*)d_in[1];  const float* b0 = (const float*)d_in[2];
  const float* W1 = (const float*)d_in[3];  const float* b1 = (const float*)d_in[4];
  const float* W2 = (const float*)d_in[5];  const float* b2 = (const float*)d_in[6];
  const float* Wp = (const float*)d_in[7];  const float* bp = (const float*)d_in[8];
  const float* Wv = (const float*)d_in[9];  const float* bv = (const float*)d_in[10];
  const float* Wb1 = (const float*)d_in[11]; const float* bb1 = (const float*)d_in[12];
  const float* Wb2 = (const float*)d_in[13]; const float* bb2 = (const float*)d_in[14];
  float* out = (float*)d_out;               // [4096*120 t_pitch | 4096*2 voice]

  // ---- workspace carve (bytes, 256-aligned) ----
  char* base = (char*)d_ws;
  size_t off = 0;
  auto take = [&](size_t bytes) -> void* {
    void* p = base + off;
    off = (off + bytes + 255) & ~(size_t)255;
    return p;
  };
  uint16_t* W0b  = (uint16_t*)take(2048UL * 1056 * 2);
  uint16_t* W1b  = (uint16_t*)take(2048UL * 2048 * 2);
  uint16_t* W2b  = (uint16_t*)take(1024UL * 2048 * 2);
  uint16_t* Wpb  = (uint16_t*)take(128UL  * 1024 * 2);
  uint16_t* Wb1b = (uint16_t*)take(256UL  * 256  * 2);
  uint16_t* Wb2b = (uint16_t*)take(256UL  * 256  * 2);
  uint16_t* X0b  = (uint16_t*)take(4096UL * 1056 * 2);
  uint16_t* H1b  = (uint16_t*)take(4096UL * 2048 * 2);
  uint16_t* H2b  = (uint16_t*)take(4096UL * 2048 * 2);
  uint16_t* H3b  = (uint16_t*)take(4096UL * 1024 * 2);
  float*    P0f  = (float*)   take(4096UL * 128  * 4);
  float*    Af   = (float*)   take(4096UL * 128  * 4);
  uint16_t* OPb  = (uint16_t*)take(2048UL * 256  * 2);
  uint16_t* Hbb  = (uint16_t*)take(2048UL * 256  * 2);
  float*    OUTf = (float*)   take(2048UL * 256  * 4);
  (void)ws_size; (void)in_sizes; (void)n_in; (void)out_size;

  auto conv = [&](const float* s, int Rin, int Cin, uint16_t* d, long Rout, int Cout) {
    long total = Rout * (long)Cout;
    int blocks = (int)((total + 255) / 256);
    convert_pad_bf16<<<blocks, 256, 0, stream>>>(s, Rin, Cin, d, total, Cout);
  };

  // ---- weight / input conversion to padded bf16 ----
  conv(W0,  2048, 1025, W0b,  2048, 1056);
  conv(W1,  2048, 2048, W1b,  2048, 2048);
  conv(W2,  1024, 2048, W2b,  1024, 2048);
  conv(Wp,   120, 1024, Wpb,   128, 1024);
  conv(Wb1,  240,  240, Wb1b,  256,  256);
  conv(Wb2,  240,  240, Wb2b,  256,  256);
  conv(feat, 4096, 1025, X0b, 4096, 1056);

  // Block tile = 64 (M) x 256 (N); waves early-out when nbase >= Npad.
  auto gemm = [&](const uint16_t* A, int lda, const uint16_t* B, int ldb,
                  const float* bias, int bias_n, uint16_t* Cb, float* Cf, int ldc,
                  int M, int Npad, int K, int act) {
    dim3 grid(M / 64, (Npad + 255) / 256);
    gemm_bf16_wmma<<<grid, 256, 0, stream>>>(A, lda, B, ldb, bias, bias_n,
                                             Cb, Cf, ldc, Npad, K, act);
  };

  // ---- MLP trunk (relu) ----
  gemm(X0b, 1056, W0b, 1056, b0, 2048, H1b, nullptr, 2048, 4096, 2048, 1056, 1);
  gemm(H1b, 2048, W1b, 2048, b1, 2048, H2b, nullptr, 2048, 4096, 2048, 2048, 1);
  gemm(H2b, 2048, W2b, 2048, b2, 1024, H3b, nullptr, 1024, 4096, 1024, 2048, 1);

  // ---- heads ----
  gemm(H3b, 1024, Wpb, 1024, bp, 120, nullptr, P0f, 128, 4096, 128, 1024, 1);
  voice_head<<<32, 256, 0, stream>>>(H3b, Wv, bv, out + 4096L * 120);

  // ---- butterfly network ----
  permute_init<<<(4096 * 128) / 256, 256, 0, stream>>>(P0f, Af);
  for (int stage = 0; stage < 8; ++stage) {
    butterfly_pack<<<(2048 * 256) / 256, 256, 0, stream>>>(Af, OPb, stage);
    gemm(OPb, 256, Wb1b, 256, bb1, 240, Hbb, nullptr, 256, 2048, 256, 256, 2);
    gemm(Hbb, 256, Wb2b, 256, bb2, 240, nullptr, OUTf, 256, 2048, 256, 256, 2);
    butterfly_unpack<<<(2048 * 256) / 256, 256, 0, stream>>>(OUTf, Af, stage);
  }

  // ---- final bit-reversal gather into t_pitch ----
  final_gather<<<4096, 128, 0, stream>>>(Af, out);
}
